// DiffDock_38087769981433
// MI455X (gfx1250) — matrix operand order repaired and checked by reading, not compiled, not run on hardware
//
#include <hip/hip_runtime.h>
#include <stdint.h>

#define NS 32
#define NV 8
#define HID 96
#define WNUM 1664
#define NNODES 40000
#define NEDGES 120000
#define TILE_E 32          // two 16-row WMMA M-tiles per block
#define OUTD 80            // 32 (out0) + 24 (out1o) + 24 (out1e)

typedef __attribute__((ext_vector_type(16))) __bf16 v16bf;
typedef __attribute__((ext_vector_type(8)))  float  v8f;

union Frag16 { v16bf v; uint4 q[2]; };
union FragD  { v8f v; float f[8]; };

struct AFrags { Frag16 a0, a1, a2; };

__device__ __forceinline__ unsigned short f2bf(float f) {
  unsigned int u = __builtin_bit_cast(unsigned int, f);
  unsigned int lsb = (u >> 16) & 1u;
  return (unsigned short)((u + 0x7FFFu + lsb) >> 16);
}

// Three K-step A fragments of a 16x96 bf16 tile (ISA wave32 A layout:
// lanes 0-15 hold row L, K in {0..7,16..23}+32s; lanes 16-31 K in {8..15,24..31}+32s)
__device__ __forceinline__ AFrags load_afrags(const unsigned short* sA, int row, int hi) {
  AFrags A;
  const int kA0 = hi * 8;
  A.a0.q[0] = *reinterpret_cast<const uint4*>(&sA[row * HID + kA0]);
  A.a0.q[1] = *reinterpret_cast<const uint4*>(&sA[row * HID + kA0 + 16]);
  A.a1.q[0] = *reinterpret_cast<const uint4*>(&sA[row * HID + 32 + kA0]);
  A.a1.q[1] = *reinterpret_cast<const uint4*>(&sA[row * HID + 32 + kA0 + 16]);
  A.a2.q[0] = *reinterpret_cast<const uint4*>(&sA[row * HID + 64 + kA0]);
  A.a2.q[1] = *reinterpret_cast<const uint4*>(&sA[row * HID + 64 + kA0 + 16]);
  return A;
}

// Two 16x16 output tiles (rows 0-15 / 16-31), K=96: each B fragment is loaded
// once and consumed by two back-to-back WMMAs (halves L2 B-traffic per flop).
__device__ __forceinline__ void wmma96x2(const AFrags& A0, const AFrags& A1,
                                         const unsigned short* __restrict__ Wt,
                                         int hi, int c, v8f& acc0, v8f& acc1) {
  const int kb = hi * 8;
  Frag16 b;
  b.q[0] = *reinterpret_cast<const uint4*>(&Wt[c * HID + kb]);
  b.q[1] = *reinterpret_cast<const uint4*>(&Wt[c * HID + kb + 16]);
  acc0 = __builtin_amdgcn_wmma_f32_16x16x32_bf16(false, A0.a0.v, false, b.v, (short)0, acc0, false, false);
  acc1 = __builtin_amdgcn_wmma_f32_16x16x32_bf16(false, A1.a0.v, false, b.v, (short)0, acc1, false, false);
  b.q[0] = *reinterpret_cast<const uint4*>(&Wt[c * HID + 32 + kb]);
  b.q[1] = *reinterpret_cast<const uint4*>(&Wt[c * HID + 32 + kb + 16]);
  acc0 = __builtin_amdgcn_wmma_f32_16x16x32_bf16(false, A0.a1.v, false, b.v, (short)0, acc0, false, false);
  acc1 = __builtin_amdgcn_wmma_f32_16x16x32_bf16(false, A1.a1.v, false, b.v, (short)0, acc1, false, false);
  b.q[0] = *reinterpret_cast<const uint4*>(&Wt[c * HID + 64 + kb]);
  b.q[1] = *reinterpret_cast<const uint4*>(&Wt[c * HID + 64 + kb + 16]);
  acc0 = __builtin_amdgcn_wmma_f32_16x16x32_bf16(false, A0.a2.v, false, b.v, (short)0, acc0, false, false);
  acc1 = __builtin_amdgcn_wmma_f32_16x16x32_bf16(false, A1.a2.v, false, b.v, (short)0, acc1, false, false);
}

// ---------------- prep kernels ----------------

__global__ void zero_f32_kernel(float* __restrict__ p, int n) {
  int i = blockIdx.x * 256 + threadIdx.x;
  if (i < n) p[i] = 0.f;
}

__global__ void convert_w1_kernel(const float* __restrict__ W1,
                                  unsigned short* __restrict__ W1t) {
  int i = blockIdx.x * 256 + threadIdx.x;
  if (i < HID * HID) {
    int c = i / HID, k = i % HID;
    W1t[c * HID + k] = f2bf(W1[k * HID + c]);
  }
}

__global__ void convert_w2_kernel(const float* __restrict__ W2,
                                  unsigned short* __restrict__ W2t) {
  int i = blockIdx.x * 256 + threadIdx.x;
  if (i < WNUM * HID) {
    int c = i / HID, k = i % HID;
    W2t[c * HID + k] = f2bf(W2[k * WNUM + c]);
  }
}

// ---------------- fused edge kernel ----------------

__global__ __launch_bounds__(256) void edge_tp_kernel(
    const float* __restrict__ node_attr,
    const float* __restrict__ edge_attr,
    const float* __restrict__ edge_sh,
    const unsigned short* __restrict__ W1t,
    const float* __restrict__ b1,
    const unsigned short* __restrict__ W2t,
    const float* __restrict__ b2,
    const int* __restrict__ edge_src,
    const int* __restrict__ edge_dst,
    float* __restrict__ out,
    float* __restrict__ cnt)
{
  __shared__ unsigned short sEA[TILE_E * HID];  // edge_attr tile, bf16
  __shared__ unsigned short sH[TILE_E * HID];   // hidden tile, bf16
  __shared__ float sAcc[TILE_E * OUTD];         // TP accumulators
  __shared__ float sP1[TILE_E * NS];            // x0 * s0 * n0
  __shared__ float sX0n[TILE_E * NS];           // x0 * n0
  __shared__ float sP3[TILE_E * NV * 3];        // x1 * s0 * n0
  __shared__ float sP4[TILE_E * NV];            // (x1 . sh1) * inv_s3 * n0
  __shared__ float sP5[TILE_E * NV * 3];        // cross(x1, sh1) * inv_s2 * n1e
  __shared__ float sSh1[TILE_E * 3];
  __shared__ int   sSrc[TILE_E];

  const int tid = threadIdx.x;
  const int wid = tid >> 5;   // wave id (wave32)
  const int L   = tid & 31;   // lane id
  const int e0  = blockIdx.x * TILE_E;

  for (int j = tid; j < TILE_E * OUTD; j += 256) sAcc[j] = 0.f;

  // ---- Phase 0a: per-edge gather + TP factor precompute ----
  if (tid < TILE_E) {
    const int e  = tid;
    const int ge = e0 + e;
    const int dst = edge_dst[ge];
    const int src = edge_src[ge];
    sSrc[e] = src;
    atomicAdd(&cnt[src], 1.0f);

    const float* na = node_attr + (size_t)dst * (NS + 3 * NV);
    const float s0   = edge_sh[ge * 4 + 0];
    const float sh1x = edge_sh[ge * 4 + 1];
    const float sh1y = edge_sh[ge * 4 + 2];
    const float sh1z = edge_sh[ge * 4 + 3];
    sSh1[e * 3 + 0] = sh1x; sSh1[e * 3 + 1] = sh1y; sSh1[e * 3 + 2] = sh1z;

    const float n0     = 0.15811388f;  // 1/sqrt(NS+NV)
    const float n1e    = 0.35355339f;  // 1/sqrt(NV)
    const float inv_s3 = 0.57735027f;
    const float inv_s2 = 0.70710678f;

    #pragma unroll
    for (int u = 0; u < NS; ++u) {
      float x0u = na[u];
      sP1[e * NS + u]  = x0u * s0 * n0;
      sX0n[e * NS + u] = x0u * n0;   // n1o == n0
    }
    #pragma unroll
    for (int u = 0; u < NV; ++u) {
      float x = na[NS + u * 3 + 0];
      float y = na[NS + u * 3 + 1];
      float z = na[NS + u * 3 + 2];
      sP4[e * NV + u] = (x * sh1x + y * sh1y + z * sh1z) * (inv_s3 * n0);
      sP3[(e * NV + u) * 3 + 0] = x * s0 * n0;
      sP3[(e * NV + u) * 3 + 1] = y * s0 * n0;
      sP3[(e * NV + u) * 3 + 2] = z * s0 * n0;
      float cx = y * sh1z - z * sh1y;
      float cy = z * sh1x - x * sh1z;
      float cz = x * sh1y - y * sh1x;
      const float c5 = inv_s2 * n1e;
      sP5[(e * NV + u) * 3 + 0] = cx * c5;
      sP5[(e * NV + u) * 3 + 1] = cy * c5;
      sP5[(e * NV + u) * 3 + 2] = cz * c5;
    }
  }

  // ---- Phase 0b: stage edge_attr tile as bf16 ----
  for (int j = tid; j < TILE_E * HID; j += 256) {
    int e = j / HID, k = j % HID;
    sEA[j] = f2bf(edge_attr[(size_t)(e0 + e) * HID + k]);
  }
  __syncthreads();

  const int row = L & 15;
  const int l15 = L & 15;
  const int hi  = L >> 4;
  const int hi8 = hi * 8;

  // ---- Phase 1: h = relu(EA @ W1 + b1); waves 0..5 cover 6 col tiles,
  //      both 16-row M-tiles per wave sharing B fragments ----
  if (wid < 6) {
    const AFrags A0 = load_afrags(sEA, row, hi);
    const AFrags A1 = load_afrags(sEA + 16 * HID, row, hi);
    const int c = wid * 16 + l15;
    v8f acc0 = {}, acc1 = {};
    wmma96x2(A0, A1, W1t, hi, c, acc0, acc1);
    FragD d0, d1; d0.v = acc0; d1.v = acc1;
    const float bias = b1[c];
    #pragma unroll
    for (int r = 0; r < 8; ++r) {
      float h0 = d0.f[r] + bias; h0 = h0 > 0.f ? h0 : 0.f;
      float h1 = d1.f[r] + bias; h1 = h1 > 0.f ? h1 : 0.f;
      sH[(r + hi8) * HID + c]      = f2bf(h0);
      sH[(16 + r + hi8) * HID + c] = f2bf(h1);
    }
  }
  __syncthreads();

  // ---- Phase 2: w = h @ W2 + b2 fused with TP consume ----
  // A fragments of h are identical for every column tile: load once (48 VGPRs).
  const AFrags A0 = load_afrags(sH, row, hi);
  const AFrags A1 = load_afrags(sH + 16 * HID, row, hi);

  // Region w1: ct in [0,64). Stride-8 ct keeps parity -> per-lane v constant:
  // accumulate out0 contributions in registers, 16 LDS atomics at the end.
  {
    float racc0[8], racc1[8];
    #pragma unroll
    for (int r = 0; r < 8; ++r) { racc0[r] = 0.f; racc1[r] = 0.f; }
    const int v = ((wid & 1) << 4) + l15;
    for (int ct = wid; ct < 64; ct += 8) {
      const int c = ct * 16 + l15;
      __builtin_prefetch(&W2t[(c + 128) * HID], 0, 1);
      v8f acc0 = {}, acc1 = {};
      wmma96x2(A0, A1, W2t, hi, c, acc0, acc1);
      FragD d0, d1; d0.v = acc0; d1.v = acc1;
      const int u = ct >> 1;
      const float bias = b2[c];
      #pragma unroll
      for (int r = 0; r < 8; ++r) {
        racc0[r] += sP1[(hi8 + r) * NS + u]      * (d0.f[r] + bias);
        racc1[r] += sP1[(16 + hi8 + r) * NS + u] * (d1.f[r] + bias);
      }
    }
    #pragma unroll
    for (int r = 0; r < 8; ++r) {
      atomicAdd(&sAcc[(hi8 + r) * OUTD + v],      racc0[r]);
      atomicAdd(&sAcc[(16 + hi8 + r) * OUTD + v], racc1[r]);
    }
  }

  // Region w2: ct in [64,80). v = l15&7 constant per lane: reduce
  // t2[v] = sum_u x0n[u]*w2[u,v] in registers, expand by sh1[m] at the end.
  {
    float t20[8], t21[8];
    #pragma unroll
    for (int r = 0; r < 8; ++r) { t20[r] = 0.f; t21[r] = 0.f; }
    const int v = l15 & 7;
    for (int ct = 64 + wid; ct < 80; ct += 8) {
      const int c = ct * 16 + l15;
      __builtin_prefetch(&W2t[(c + 128) * HID], 0, 1);
      v8f acc0 = {}, acc1 = {};
      wmma96x2(A0, A1, W2t, hi, c, acc0, acc1);
      FragD d0, d1; d0.v = acc0; d1.v = acc1;
      const int u = ((ct - 64) << 1) + (l15 >> 3);
      const float bias = b2[c];
      #pragma unroll
      for (int r = 0; r < 8; ++r) {
        t20[r] += sX0n[(hi8 + r) * NS + u]      * (d0.f[r] + bias);
        t21[r] += sX0n[(16 + hi8 + r) * NS + u] * (d1.f[r] + bias);
      }
    }
    #pragma unroll
    for (int r = 0; r < 8; ++r) {
      const int ea = hi8 + r, eb = 16 + hi8 + r;
      atomicAdd(&sAcc[ea * OUTD + 32 + v * 3 + 0], t20[r] * sSh1[ea * 3 + 0]);
      atomicAdd(&sAcc[ea * OUTD + 32 + v * 3 + 1], t20[r] * sSh1[ea * 3 + 1]);
      atomicAdd(&sAcc[ea * OUTD + 32 + v * 3 + 2], t20[r] * sSh1[ea * 3 + 2]);
      atomicAdd(&sAcc[eb * OUTD + 32 + v * 3 + 0], t21[r] * sSh1[eb * 3 + 0]);
      atomicAdd(&sAcc[eb * OUTD + 32 + v * 3 + 1], t21[r] * sSh1[eb * 3 + 1]);
      atomicAdd(&sAcc[eb * OUTD + 32 + v * 3 + 2], t21[r] * sSh1[eb * 3 + 2]);
    }
  }

  // Region w3: ct in [80,84) -> waves 0..3 (one tile each), wave-uniform branch.
  if (wid < 4) {
    const int ct = 80 + wid;
    const int c = ct * 16 + l15;
    v8f acc0 = {}, acc1 = {};
    wmma96x2(A0, A1, W2t, hi, c, acc0, acc1);
    FragD d0, d1; d0.v = acc0; d1.v = acc1;
    const int u = ((ct - 80) << 1) + (l15 >> 3);
    const int v = l15 & 7;
    const float bias = b2[c];
    #pragma unroll
    for (int r = 0; r < 8; ++r) {
      const int ea = hi8 + r, eb = 16 + hi8 + r;
      const float wa = d0.f[r] + bias, wb = d1.f[r] + bias;
      atomicAdd(&sAcc[ea * OUTD + 32 + v * 3 + 0], sP3[(ea * NV + u) * 3 + 0] * wa);
      atomicAdd(&sAcc[ea * OUTD + 32 + v * 3 + 1], sP3[(ea * NV + u) * 3 + 1] * wa);
      atomicAdd(&sAcc[ea * OUTD + 32 + v * 3 + 2], sP3[(ea * NV + u) * 3 + 2] * wa);
      atomicAdd(&sAcc[eb * OUTD + 32 + v * 3 + 0], sP3[(eb * NV + u) * 3 + 0] * wb);
      atomicAdd(&sAcc[eb * OUTD + 32 + v * 3 + 1], sP3[(eb * NV + u) * 3 + 1] * wb);
      atomicAdd(&sAcc[eb * OUTD + 32 + v * 3 + 2], sP3[(eb * NV + u) * 3 + 2] * wb);
    }
  }

  // Region w4: ct in [84,100). Two stride-8 tiles per wave, same parity ->
  // per-lane v constant: register accumulation again.
  {
    float racc0[8], racc1[8];
    #pragma unroll
    for (int r = 0; r < 8; ++r) { racc0[r] = 0.f; racc1[r] = 0.f; }
    const int ct0 = 84 + ((wid + 4) & 7);
    const int v = ((ct0 & 1) << 4) + l15;
    for (int ct = ct0; ct < 100; ct += 8) {
      const int c = ct * 16 + l15;
      if (c + 128 < WNUM) __builtin_prefetch(&W2t[(c + 128) * HID], 0, 1);
      v8f acc0 = {}, acc1 = {};
      wmma96x2(A0, A1, W2t, hi, c, acc0, acc1);
      FragD d0, d1; d0.v = acc0; d1.v = acc1;
      const int u = (ct - 84) >> 1;
      const float bias = b2[c];
      #pragma unroll
      for (int r = 0; r < 8; ++r) {
        racc0[r] += sP4[(hi8 + r) * NV + u]      * (d0.f[r] + bias);
        racc1[r] += sP4[(16 + hi8 + r) * NV + u] * (d1.f[r] + bias);
      }
    }
    #pragma unroll
    for (int r = 0; r < 8; ++r) {
      atomicAdd(&sAcc[(hi8 + r) * OUTD + v],      racc0[r]);
      atomicAdd(&sAcc[(16 + hi8 + r) * OUTD + v], racc1[r]);
    }
  }

  // Region w5: ct in [100,104) -> waves 4..7 (one tile each), wave-uniform branch.
  if (wid >= 4) {
    const int ct = 96 + wid;   // 100..103
    const int c = ct * 16 + l15;
    v8f acc0 = {}, acc1 = {};
    wmma96x2(A0, A1, W2t, hi, c, acc0, acc1);
    FragD d0, d1; d0.v = acc0; d1.v = acc1;
    const int u = ((ct - 100) << 1) + (l15 >> 3);
    const int v = l15 & 7;
    const float bias = b2[c];
    #pragma unroll
    for (int r = 0; r < 8; ++r) {
      const int ea = hi8 + r, eb = 16 + hi8 + r;
      const float wa = d0.f[r] + bias, wb = d1.f[r] + bias;
      atomicAdd(&sAcc[ea * OUTD + 56 + v * 3 + 0], sP5[(ea * NV + u) * 3 + 0] * wa);
      atomicAdd(&sAcc[ea * OUTD + 56 + v * 3 + 1], sP5[(ea * NV + u) * 3 + 1] * wa);
      atomicAdd(&sAcc[ea * OUTD + 56 + v * 3 + 2], sP5[(ea * NV + u) * 3 + 2] * wa);
      atomicAdd(&sAcc[eb * OUTD + 56 + v * 3 + 0], sP5[(eb * NV + u) * 3 + 0] * wb);
      atomicAdd(&sAcc[eb * OUTD + 56 + v * 3 + 1], sP5[(eb * NV + u) * 3 + 1] * wb);
      atomicAdd(&sAcc[eb * OUTD + 56 + v * 3 + 2], sP5[(eb * NV + u) * 3 + 2] * wb);
    }
  }
  __syncthreads();

  // ---- Phase 3: scatter-add to nodes ----
  for (int j = tid; j < TILE_E * OUTD; j += 256) {
    int e = j / OUTD, k = j % OUTD;
    atomicAdd(&out[(size_t)sSrc[e] * OUTD + k], sAcc[j]);
  }
}

// ---------------- finalize: segment mean ----------------

__global__ void finalize_kernel(float* __restrict__ out, const float* __restrict__ cnt) {
  int i = blockIdx.x * 256 + threadIdx.x;
  if (i < NNODES * OUTD) {
    int n = i / OUTD;
    float c = cnt[n];
    out[i] *= 1.0f / (c > 1.0f ? c : 1.0f);
  }
}

// ---------------- launch ----------------

extern "C" void kernel_launch(void* const* d_in, const int* in_sizes, int n_in,
                              void* d_out, int out_size, void* d_ws, size_t ws_size,
                              hipStream_t stream) {
  const float* node_attr = (const float*)d_in[0];
  const float* edge_attr = (const float*)d_in[1];
  const float* edge_sh   = (const float*)d_in[2];
  const float* W1        = (const float*)d_in[3];
  const float* b1        = (const float*)d_in[4];
  const float* W2        = (const float*)d_in[5];
  const float* b2        = (const float*)d_in[6];
  const int*   edge_src  = (const int*)d_in[7];
  const int*   edge_dst  = (const int*)d_in[8];
  float* out = (float*)d_out;

  // workspace: bf16 W1t (18KB) | bf16 W2t (312KB, L2-resident) | cnt (156KB)
  unsigned short* W1t = (unsigned short*)d_ws;
  unsigned short* W2t = W1t + HID * HID;
  float* cnt = (float*)(void*)(W2t + (size_t)WNUM * HID);

  zero_f32_kernel<<<(NNODES * OUTD + 255) / 256, 256, 0, stream>>>(out, NNODES * OUTD);
  zero_f32_kernel<<<(NNODES + 255) / 256, 256, 0, stream>>>(cnt, NNODES);
  convert_w1_kernel<<<(HID * HID + 255) / 256, 256, 0, stream>>>(W1, W1t);
  convert_w2_kernel<<<(WNUM * HID + 255) / 256, 256, 0, stream>>>(W2, W2t);

  edge_tp_kernel<<<NEDGES / TILE_E, 256, 0, stream>>>(
      node_attr, edge_attr, edge_sh, W1t, b1, W2t, b2,
      edge_src, edge_dst, out, cnt);

  finalize_kernel<<<(NNODES * OUTD + 255) / 256, 256, 0, stream>>>(out, cnt);
}
